// SelfAttention_20968030339358
// MI455X (gfx1250) — matrix operand order repaired and checked by reading, not compiled, not run on hardware
//
#include <hip/hip_runtime.h>

typedef __attribute__((ext_vector_type(16))) _Float16 v16h;
typedef __attribute__((ext_vector_type(8)))  _Float16 v8h;
typedef __attribute__((ext_vector_type(8)))  float    v8f;

#define EMBED 1024
#define HEADS 16
#define HDIM  64
#define SEQ   2048
#define NBATCH 2

#if __has_builtin(__builtin_amdgcn_sched_barrier)
#define SCHED_FENCE() __builtin_amdgcn_sched_barrier(0)
#else
#define SCHED_FENCE() asm volatile("" ::: "memory")
#endif

__device__ __forceinline__ v8f wmma16(v16h a, v16h b, v8f c) {
  // D = A(16x32 f16) * B(32x16 f16) + C(16x16 f32)
  return __builtin_amdgcn_wmma_f32_16x16x32_f16(
      /*neg_a=*/false, a, /*neg_b=*/false, b,
      /*c_mod=*/(short)0, c, /*reuse_a=*/false, /*reuse_b=*/false);
}

// A-fragment (16x32, 16-bit): lane holds row (lane&15); slots 0-15 = K
// 2*koff..2*koff+15, koff = 8*(lane>=16).
__device__ __forceinline__ v16h load_afrag(const _Float16* base, int koff) {
  v8h lo = *(const v8h*)(base + 2 * koff);
  v8h hi = *(const v8h*)(base + 2 * koff + 8);
  v16h a;
#pragma unroll
  for (int s = 0; s < 8; ++s) { a[s] = lo[s]; a[s + 8] = hi[s]; }
  return a;
}

__global__ void cvt_f32_f16(const float* __restrict__ src,
                            _Float16* __restrict__ dst, int n) {
  int i = blockIdx.x * blockDim.x + threadIdx.x;
  int stride = gridDim.x * blockDim.x;
  for (; i < n; i += stride) dst[i] = (_Float16)src[i];
}

// out = A[M,1024] @ Bw[1024,1024]^T (torch Linear). Each wave: 16(m) x 64(n).
// K loop: fully unrolled, two-stage buffered, with sched_barrier pinning the
// pipeline shape: step i+1's loads stay ABOVE step i's WMMAs, so each WMMA
// group waits only on loads issued one stage earlier (non-zero loadcnt).
// mode 0: f16 out, row-major [M,1024]
// mode 1: f16 out, transposed-packed vT[(b*1024+e)*2048 + seq], m = b*2048+seq
// mode 2: f32 out, row-major, + bias
__global__ __launch_bounds__(256) void gemm_xwt(
    const _Float16* __restrict__ A, const _Float16* __restrict__ Bw,
    void* __restrict__ out, const float* __restrict__ bias, int M, int mode) {
  const int lane = threadIdx.x & 31;
  const int half = lane >> 4;
  const int col  = lane & 15;
  const int wave = blockIdx.x * (blockDim.x >> 5) + (threadIdx.x >> 5);
  const int mtiles = M >> 4;
  if (wave >= mtiles * 16) return;
  const int m0 = (wave >> 4) << 4;       // 16-row tile
  const int n0 = (wave & 15) << 6;       // 64-col tile

  const _Float16* arow = A + (size_t)(m0 + col) * EMBED;
  const _Float16* brow[4];
#pragma unroll
  for (int t = 0; t < 4; ++t)
    brow[t] = Bw + (size_t)(n0 + 16 * t + col) * EMBED + 16 * half;

  v8f acc[4] = {};
  // stage buffer for kk = 0
  v16h aCur = load_afrag(arow, 8 * half);
  v16h bCur[4];
#pragma unroll
  for (int t = 0; t < 4; ++t) bCur[t] = *(const v16h*)(brow[t]);

#pragma unroll
  for (int kk0 = 0; kk0 < EMBED; kk0 += 32) {
    v16h aNxt, bNxt[4];
    if (kk0 + 32 < EMBED) {  // issue next-step loads BEFORE this step's WMMAs
      aNxt = load_afrag(arow + kk0 + 32, 8 * half);
#pragma unroll
      for (int t = 0; t < 4; ++t) bNxt[t] = *(const v16h*)(brow[t] + kk0 + 32);
    }
    SCHED_FENCE();  // keep the loads above the WMMAs through scheduling
#pragma unroll
    for (int t = 0; t < 4; ++t) acc[t] = wmma16(aCur, bCur[t], acc[t]);
    aCur = aNxt;  // pure SSA rename after full unroll
#pragma unroll
    for (int t = 0; t < 4; ++t) bCur[t] = bNxt[t];
  }

  if (mode == 0) {
    _Float16* o16 = (_Float16*)out;
#pragma unroll
    for (int t = 0; t < 4; ++t)
#pragma unroll
      for (int r = 0; r < 8; ++r)
        o16[(size_t)(m0 + r + 8 * half) * EMBED + n0 + 16 * t + col] =
            (_Float16)acc[t][r];
  } else if (mode == 1) {
    _Float16* o16 = (_Float16*)out;
    const int bb = m0 >> 11;                 // batch (seq = 2048)
    const int seq0 = (m0 & 2047) + 8 * half; // 8 consecutive seq positions
#pragma unroll
    for (int t = 0; t < 4; ++t) {
      v8h pk;
#pragma unroll
      for (int r = 0; r < 8; ++r) pk[r] = (_Float16)acc[t][r];
      *(v8h*)(o16 + ((size_t)bb * EMBED + n0 + 16 * t + col) * SEQ + seq0) = pk;
    }
  } else {
    float* of = (float*)out;
#pragma unroll
    for (int t = 0; t < 4; ++t) {
      float bv = bias[n0 + 16 * t + col];
#pragma unroll
      for (int r = 0; r < 8; ++r)
        of[(size_t)(m0 + r + 8 * half) * EMBED + n0 + 16 * t + col] =
            acc[t][r] + bv;
    }
  }
}

// Flash-attention: one wave per (batch, head, 16-row q tile).
// Key chunks of 32 are double-buffered (ping-pong fragment sets, j0 step 64,
// last chunk pair peeled): one buffer's K/V/mask global loads fly while the
// other buffer runs energy-WMMA -> online softmax -> LDS P tile -> PV-WMMA.
__global__ __launch_bounds__(256) void attn_kernel(
    const _Float16* __restrict__ qp, const _Float16* __restrict__ kp,
    const _Float16* __restrict__ vT, const int* __restrict__ mask,
    _Float16* __restrict__ ctx) {
  __shared__ _Float16 lds[8][16 * 32];  // per-wave 16x32 P tile
  const int lane = threadIdx.x & 31;
  const int half = lane >> 4;
  const int col  = lane & 15;
  const int w    = threadIdx.x >> 5;
  const int task = blockIdx.x * 8 + w;   // 4096 tasks total
  const int b  = task >> 11;             // 2048 tasks per batch
  const int h  = (task >> 7) & 15;
  const int q0 = (task & 127) << 4;

  // Q fragments (loop invariant): rows q0..q0+15, d split 0..31 / 32..63
  const _Float16* qbase =
      qp + (size_t)(b * SEQ + q0 + col) * EMBED + h * HDIM;
  v16h qa0 = load_afrag(qbase, 8 * half);
  v16h qa1 = load_afrag(qbase + 32, 8 * half);

  float m_run[8], l_run[8];
  v8f o[4] = {};
#pragma unroll
  for (int r = 0; r < 8; ++r) { m_run[r] = -3.0e38f; l_run[r] = 0.0f; }
  const float scale = 0.03125f;  // 1/sqrt(EMBED)

  v16h kf0[2][2], kf1[2][2], vf[2][4];
  int mk[2][2][8];

  auto load_chunk = [&](int buf, int j0) {
#pragma unroll
    for (int c = 0; c < 2; ++c) {
      const _Float16* kb =
          kp + (size_t)(b * SEQ + j0 + 16 * c + col) * EMBED + h * HDIM;
      kf0[buf][c] = *(const v16h*)(kb + 16 * half);       // d 0..31
      kf1[buf][c] = *(const v16h*)(kb + 32 + 16 * half);  // d 32..63
      const int* mrow =
          mask + ((size_t)b * SEQ + q0 + 8 * half) * SEQ + j0 + 16 * c + col;
#pragma unroll
      for (int r = 0; r < 8; ++r) mk[buf][c][r] = mrow[(size_t)r * SEQ];
    }
#pragma unroll
    for (int t = 0; t < 4; ++t) {
      // B(k,n) = V[j0+k][h*64+16t+n]; vT is [b][e][seq] so k is contiguous
      const _Float16* vb =
          vT + ((size_t)b * EMBED + h * HDIM + 16 * t + col) * SEQ + j0 +
          16 * half;
      vf[buf][t] = *(const v16h*)(vb);
    }
  };

  auto process_chunk = [&](int buf) {
    // energy: 16q x 32keys, reduce over d=64 (2 chained WMMAs per half-tile)
    float s0[8], s1[8];
#pragma unroll
    for (int c = 0; c < 2; ++c) {
      v8f e = {};
      e = wmma16(qa0, kf0[buf][c], e);
      e = wmma16(qa1, kf1[buf][c], e);
      float* sd = c ? s1 : s0;
#pragma unroll
      for (int r = 0; r < 8; ++r)
        sd[r] = mk[buf][c][r] ? e[r] * scale : -1.0e30f;
    }
    // online softmax: each row's 16 values live in one 16-lane group
#pragma unroll
    for (int r = 0; r < 8; ++r) {
      float vmax = fmaxf(s0[r], s1[r]);
      vmax = fmaxf(vmax, __shfl_xor(vmax, 1));
      vmax = fmaxf(vmax, __shfl_xor(vmax, 2));
      vmax = fmaxf(vmax, __shfl_xor(vmax, 4));
      vmax = fmaxf(vmax, __shfl_xor(vmax, 8));
      float mn = fmaxf(m_run[r], vmax);
      float alpha = __expf(m_run[r] - mn);
      float p0 = __expf(s0[r] - mn);
      float p1 = __expf(s1[r] - mn);
      float rs = p0 + p1;
      rs += __shfl_xor(rs, 1);
      rs += __shfl_xor(rs, 2);
      rs += __shfl_xor(rs, 4);
      rs += __shfl_xor(rs, 8);
      l_run[r] = l_run[r] * alpha + rs;
      m_run[r] = mn;
#pragma unroll
      for (int t = 0; t < 4; ++t) o[t][r] *= alpha;
      // C-layout -> LDS at [m][key_local] (within-wave DS is in-order)
      lds[w][(r + 8 * half) * 32 + col]      = (_Float16)p0;
      lds[w][(r + 8 * half) * 32 + 16 + col] = (_Float16)p1;
    }
    // Re-load P as an A-fragment (K = 32 local keys)
    v16h pa = load_afrag(&lds[w][col * 32], 8 * half);
#pragma unroll
    for (int t = 0; t < 4; ++t) o[t] = wmma16(pa, vf[buf][t], o[t]);
  };

  load_chunk(0, 0);
  for (int j0 = 0; j0 < SEQ - 64; j0 += 64) {
    load_chunk(1, j0 + 32);
    process_chunk(0);
    load_chunk(0, j0 + 64);
    process_chunk(1);
  }
  // peeled final chunk pair: keys SEQ-64..SEQ-33 and SEQ-32..SEQ-1
  load_chunk(1, SEQ - 32);
  process_chunk(0);
  process_chunk(1);

#pragma unroll
  for (int r = 0; r < 8; ++r) {
    float inv = 1.0f / fmaxf(l_run[r], 1e-30f);
#pragma unroll
    for (int t = 0; t < 4; ++t)
      ctx[(size_t)(b * SEQ + q0 + r + 8 * half) * EMBED + h * HDIM + 16 * t +
          col] = (_Float16)(o[t][r] * inv);
  }
}

extern "C" void kernel_launch(void* const* d_in, const int* in_sizes, int n_in,
                              void* d_out, int out_size, void* d_ws,
                              size_t ws_size, hipStream_t stream) {
  const float* queries = (const float*)d_in[0];
  const float* keysp   = (const float*)d_in[1];
  const float* values  = (const float*)d_in[2];
  const int*   mask    = (const int*)d_in[3];
  const float* Wq = (const float*)d_in[4];
  const float* Wk = (const float*)d_in[5];
  const float* Wv = (const float*)d_in[6];
  const float* Wo = (const float*)d_in[7];
  const float* bo = (const float*)d_in[8];

  const int IN_ELEMS = NBATCH * SEQ * EMBED;   // 4,194,304
  const int W_ELEMS  = EMBED * EMBED;          // 1,048,576
  const size_t MB = 1u << 20;

  char* ws = (char*)d_ws;
  _Float16* qin16   = (_Float16*)(ws + 0 * MB);
  _Float16* kin16   = (_Float16*)(ws + 8 * MB);
  _Float16* vin16   = (_Float16*)(ws + 16 * MB);
  _Float16* Wq16    = (_Float16*)(ws + 24 * MB);
  _Float16* Wk16    = (_Float16*)(ws + 26 * MB);
  _Float16* Wv16    = (_Float16*)(ws + 28 * MB);
  _Float16* Wo16    = (_Float16*)(ws + 30 * MB);
  _Float16* qproj16 = (_Float16*)(ws + 32 * MB);
  _Float16* kproj16 = (_Float16*)(ws + 40 * MB);
  _Float16* vT16    = (_Float16*)(ws + 48 * MB);
  _Float16* ctx16   = (_Float16*)(ws + 56 * MB);

  cvt_f32_f16<<<512, 256, 0, stream>>>(queries, qin16, IN_ELEMS);
  cvt_f32_f16<<<512, 256, 0, stream>>>(keysp,   kin16, IN_ELEMS);
  cvt_f32_f16<<<512, 256, 0, stream>>>(values,  vin16, IN_ELEMS);
  cvt_f32_f16<<<256, 256, 0, stream>>>(Wq, Wq16, W_ELEMS);
  cvt_f32_f16<<<256, 256, 0, stream>>>(Wk, Wk16, W_ELEMS);
  cvt_f32_f16<<<256, 256, 0, stream>>>(Wv, Wv16, W_ELEMS);
  cvt_f32_f16<<<256, 256, 0, stream>>>(Wo, Wo16, W_ELEMS);

  const int M = NBATCH * SEQ;  // 4096
  const int gemm_blocks = (M / 16) * 16 / 8;  // 8 waves per 256-thread block
  gemm_xwt<<<gemm_blocks, 256, 0, stream>>>(qin16, Wq16, qproj16, nullptr, M, 0);
  gemm_xwt<<<gemm_blocks, 256, 0, stream>>>(kin16, Wk16, kproj16, nullptr, M, 0);
  gemm_xwt<<<gemm_blocks, 256, 0, stream>>>(vin16, Wv16, vT16,    nullptr, M, 1);

  attn_kernel<<<512, 256, 0, stream>>>(qproj16, kproj16, vT16, mask, ctx16);

  gemm_xwt<<<gemm_blocks, 256, 0, stream>>>(ctx16, Wo16, d_out, bo, M, 2);
}
